// Decoder_76321568850450
// MI455X (gfx1250) — compile-verified, matrix operand order checked by reading, not run on hardware
//
#include <hip/hip_runtime.h>

// ---------------------------------------------------------------------------
// Types for CDNA5 WMMA
// ---------------------------------------------------------------------------
typedef __attribute__((ext_vector_type(16))) __bf16 v16bf;
typedef __attribute__((ext_vector_type(8)))  __bf16 v8bf;   // 16-byte load unit
typedef __attribute__((ext_vector_type(8)))  float  v8f;

// Problem constants
#define B_      4
#define C_      128
#define H_      88
#define W_      88
#define HW_     (H_*W_)          // 7744
#define CHW_    (C_*HW_)         // 991232
#define NPIX    (B_*HW_)         // 30976 (== tokens T)
#define NTOK    30976
#define HID2    1024             // 2*hidden
#define HID     512

// ---------------------------------------------------------------------------
// small helpers
// ---------------------------------------------------------------------------
__device__ __forceinline__ __bf16 f2bf(float x) {
  union { float f; unsigned u; } v; v.f = x;
  unsigned r = v.u + 0x7FFFu + ((v.u >> 16) & 1u);
  union { unsigned short s; __bf16 b; } o; o.s = (unsigned short)(r >> 16);
  return o.b;
}
__device__ __forceinline__ float bf2f(__bf16 h) {
  union { unsigned short s; __bf16 b; } i; i.b = h;
  union { unsigned u; float f; } o; o.u = ((unsigned)i.s) << 16;
  return o.f;
}
// bilinear (half-pixel, edge clamp) coefficients for 44 -> 88 upsample
__device__ __forceinline__ void blco(int d, int& s0, int& s1, float& w1) {
  float s  = 0.5f * (float)d - 0.25f;
  float fl = floorf(s);
  w1 = s - fl;
  s0 = (int)fl; s1 = s0 + 1;
  s0 = s0 < 0 ? 0 : (s0 > 43 ? 43 : s0);
  s1 = s1 < 0 ? 0 : (s1 > 43 ? 43 : s1);
}

// Load one 16x32 bf16 operand fragment per ISA VGPR layout:
// lane L: row = L&15 ; K-halves {kb..kb+7, 16+kb..16+kb+7}, kb = 8*(L>=16)
template <typename P>
__device__ __forceinline__ v16bf load_frag(P base, int ld, int lane) {
  int row = lane & 15;
  int kb  = (lane >> 4) << 3;
  auto p  = base + (long)row * ld + kb;
  v8bf lo = *(const v8bf*)(p);
  v8bf hi = *(const v8bf*)(p + 16);
  v16bf f;
#pragma unroll
  for (int i = 0; i < 8; ++i) { f[i] = lo[i]; f[i + 8] = hi[i]; }
  return f;
}

// ---------------------------------------------------------------------------
// 1) f32 -> bf16 weight conversion
// ---------------------------------------------------------------------------
__global__ void k_w2bf(const float* __restrict__ s, __bf16* __restrict__ d, int n) {
  int i = blockIdx.x * 256 + threadIdx.x;
  if (i < n) d[i] = f2bf(s[i]);
}

// ---------------------------------------------------------------------------
// 2) LN1 + bilinear(mask, edge) + window tokenize  -> Aq, Akv (bf16, T x 128)
// ---------------------------------------------------------------------------
__global__ void k_pre(const float* __restrict__ x, const float* __restrict__ mask,
                      const float* __restrict__ edge, const float* __restrict__ g1,
                      const float* __restrict__ b1,
                      __bf16* __restrict__ Aq, __bf16* __restrict__ Akv) {
  int pix = blockIdx.x * 256 + threadIdx.x;
  if (pix >= NPIX) return;
  int b = pix / HW_, rem = pix % HW_;
  int H = rem / W_, W = rem % W_;
  const float* xb = x + (long)b * CHW_ + rem;

  float s = 0.f, ss = 0.f;
#pragma unroll 4
  for (int c = 0; c < C_; ++c) { float v = xb[c * HW_]; s += v; ss += v * v; }
  float mu  = s * (1.f / C_);
  float var = ss * (1.f / C_) - mu * mu;
  float inv = rsqrtf(var + 1e-5f);

  int y0, y1, x0, x1; float wy, wx;
  blco(H, y0, y1, wy); blco(W, x0, x1, wx);
  const float* mb = mask + (long)b * 44 * 44;
  float m = (1.f - wy) * ((1.f - wx) * mb[y0 * 44 + x0] + wx * mb[y0 * 44 + x1]) +
                   wy  * ((1.f - wx) * mb[y1 * 44 + x0] + wx * mb[y1 * 44 + x1]);

  int hb = H >> 2, wb = W >> 2, wi = H & 3, wj = W & 3;
  int tk = ((b * 22 + hb) * 22 + wb) * 16 + wi * 4 + wj;
  __bf16* aq  = Aq  + (long)tk * C_;
  __bf16* akv = Akv + (long)tk * C_;
  const float* eb = edge + (long)b * C_ * 44 * 44;

  for (int c = 0; c < C_; ++c) {
    float xv = (xb[c * HW_] - mu) * inv * g1[c] + b1[c];
    akv[c] = f2bf(xv * m);
    const float* ec = eb + (long)c * 44 * 44;
    float e = (1.f - wy) * ((1.f - wx) * ec[y0 * 44 + x0] + wx * ec[y0 * 44 + x1]) +
                     wy  * ((1.f - wx) * ec[y1 * 44 + x0] + wx * ec[y1 * 44 + x1]);
    aq[c] = f2bf(e);
  }
}

// ---------------------------------------------------------------------------
// 3) Generic bf16 WMMA GEMM:  D(MxN,f32) = A(MxK) * B(NxK)^T
//    block = 256 thr (8 waves), block tile 128(M) x 128(N), K chunked by 128
//    B chunk staged in LDS via GLOBAL_LOAD_ASYNC_TO_LDS_B128 (ASYNCcnt path);
//    modes: 0 = f32 store, 1 = bf16 store, 2 = transpose-to-NCHW + residual
// ---------------------------------------------------------------------------
__global__ void k_gemm(const __bf16* __restrict__ A, const __bf16* __restrict__ B,
                       int K, int N, int mode,
                       float* __restrict__ outf, __bf16* __restrict__ outb,
                       const float* __restrict__ resid) {
  __shared__ alignas(16) __bf16 Blds[128 * 128];
  int tid  = threadIdx.x;
  int lane = tid & 31, wv = tid >> 5;
  int m0    = blockIdx.x * 128 + wv * 16;
  int nbase = blockIdx.y * 128;

  v8f acc[8];
#pragma unroll
  for (int i = 0; i < 8; ++i)
#pragma unroll
    for (int j = 0; j < 8; ++j) acc[i][j] = 0.f;

  for (int k0 = 0; k0 < K; k0 += 128) {
    __syncthreads();
    // async DMA stage: global -> LDS without touching VGPRs (ASYNCcnt).
    // LDS offset is derived from &Blds[...] (low 32 bits of its flat address
    // == LDS byte offset, ISA 10.2), which also makes Blds escape so the
    // compiler keeps the LDS data flow (ds_loads below) alive.
    for (int i = tid; i < 2048; i += 256) {          // 128x128 bf16 = 2048 x 16B
      int r = i >> 4, cv = i & 15;
      const char* gaddr = (const char*)&B[(long)(nbase + r) * K + k0 + cv * 8];
      __bf16* lptr = &Blds[i * 8];
      unsigned lds_off = (unsigned)(unsigned long long)lptr;
      asm volatile("global_load_async_to_lds_b128 %0, %1, off"
                   :: "v"(lds_off), "v"(gaddr), "v"(lptr) : "memory");
    }
    // prefetch next A chunk while DMA is in flight
    if (k0 + 128 < K)
      __builtin_prefetch(A + (long)m0 * K + (k0 + 128), 0, 1);
    asm volatile("s_wait_asynccnt 0x0" ::: "memory");
    __syncthreads();

#pragma unroll
    for (int kk = 0; kk < 128; kk += 32) {
      v16bf af = load_frag(A + (long)m0 * K + k0 + kk, K, lane);
#pragma unroll
      for (int nt = 0; nt < 8; ++nt) {
        v16bf bf = load_frag(&Blds[nt * 16 * 128 + kk], 128, lane);
        acc[nt] = __builtin_amdgcn_wmma_f32_16x16x32_bf16(
            false, af, false, bf, (short)0, acc[nt], false, false);
      }
    }
  }

  int mrow = (lane >> 4) * 8, nloc = lane & 15;
#pragma unroll
  for (int nt = 0; nt < 8; ++nt) {
    int col = nbase + nt * 16 + nloc;
#pragma unroll
    for (int v = 0; v < 8; ++v) {
      int m = m0 + mrow + v;
      float val = acc[nt][v];
      if (mode == 0) {
        outf[(long)m * N + col] = val;
      } else if (mode == 1) {
        outb[(long)m * N + col] = f2bf(val);
      } else {
        int b = m / HW_, rem = m % HW_;
        long dst = (long)b * CHW_ + (long)col * HW_ + rem;
        outf[dst] = resid[dst] + val;
      }
    }
  }
}

// ---------------------------------------------------------------------------
// 4) Attention core: per token (64 thr) dots(64x64,K=2) -> softmax -> apply,
//    writes scrambled residual x_res = x + out
// ---------------------------------------------------------------------------
__global__ void k_attn(const float* __restrict__ Q, const float* __restrict__ Kf,
                       const float* __restrict__ Vf, const float* __restrict__ x,
                       float* __restrict__ xres) {
  int t = blockIdx.x, tid = threadIdx.x;          // 64 threads
  __shared__ float kb[128], vb[128];
  kb[tid]      = Kf[(long)t * 128 + tid];
  kb[tid + 64] = Kf[(long)t * 128 + 64 + tid];
  vb[tid]      = Vf[(long)t * 128 + tid];
  vb[tid + 64] = Vf[(long)t * 128 + 64 + tid];
  __syncthreads();

  float q0 = Q[(long)t * 128 + tid];
  float q1 = Q[(long)t * 128 + 64 + tid];
  float d[64];
  float mx = -1e30f;
#pragma unroll 8
  for (int k = 0; k < 64; ++k) {
    float dv = (q0 * kb[k] + q1 * kb[64 + k]) * 0.125f;  // dh^-0.5, dh=64
    d[k] = dv; mx = fmaxf(mx, dv);
  }
  float sum = 0.f;
#pragma unroll 8
  for (int k = 0; k < 64; ++k) { d[k] = __expf(d[k] - mx); sum += d[k]; }
  float o0 = 0.f, o1 = 0.f;
#pragma unroll 8
  for (int k = 0; k < 64; ++k) { o0 += d[k] * vb[k]; o1 += d[k] * vb[64 + k]; }
  float r = 1.f / sum; o0 *= r; o1 *= r;

  // scrambled residual placement (faithful to source reshape/transpose)
  int bw = t >> 4, n = t & 15;
  int b = bw / 484, rb = bw % 484, hb = rb / 22, wb = rb % 22;
  long base = (long)b * CHW_;
  long f0 = (long)(n * 128 + tid)      * 484 + hb * 22 + wb;
  long f1 = (long)(n * 128 + tid + 64) * 484 + hb * 22 + wb;
  xres[base + f0] = x[base + f0] + o0;
  xres[base + f1] = x[base + f1] + o1;
}

// ---------------------------------------------------------------------------
// 5) LN2 -> pixel-major bf16 (GEMM A operand)
// ---------------------------------------------------------------------------
__global__ void k_ln2(const float* __restrict__ xr, const float* __restrict__ g2,
                      const float* __restrict__ b2, __bf16* __restrict__ xn2) {
  int pix = blockIdx.x * 256 + threadIdx.x;
  if (pix >= NPIX) return;
  int b = pix / HW_, rem = pix % HW_;
  const float* xb = xr + (long)b * CHW_ + rem;
  float s = 0.f, ss = 0.f;
#pragma unroll 4
  for (int c = 0; c < C_; ++c) { float v = xb[c * HW_]; s += v; ss += v * v; }
  float mu  = s * (1.f / C_);
  float inv = rsqrtf(ss * (1.f / C_) - mu * mu + 1e-5f);
  __bf16* o = xn2 + (long)pix * C_;
  for (int c = 0; c < C_; ++c)
    o[c] = f2bf((xb[c * HW_] - mu) * inv * g2[c] + b2[c]);
}

// ---------------------------------------------------------------------------
// 6) depthwise 3x3 (zero pad) + exact-GELU gate  -> g bf16 (P x 512)
// ---------------------------------------------------------------------------
__global__ void k_dwgate(const __bf16* __restrict__ t, const float* __restrict__ wdw,
                         __bf16* __restrict__ g) {
  long idx = (long)blockIdx.x * 256 + threadIdx.x;
  if (idx >= (long)NPIX * HID) return;
  int ch = (int)(idx & (HID - 1));
  int p  = (int)(idx >> 9);
  int b = p / HW_, rem = p % HW_, H = rem / W_, W = rem % W_;
  const float* w1 = wdw + (long)ch * 9;
  const float* w2 = wdw + (long)(ch + HID) * 9;
  float d1 = 0.f, d2 = 0.f;
#pragma unroll
  for (int dy = -1; dy <= 1; ++dy) {
    int Hn = H + dy; if (Hn < 0 || Hn >= H_) continue;
#pragma unroll
    for (int dx = -1; dx <= 1; ++dx) {
      int Wn = W + dx; if (Wn < 0 || Wn >= W_) continue;
      long pp = ((long)b * HW_ + Hn * W_ + Wn) * HID2;
      int wi = (dy + 1) * 3 + (dx + 1);
      d1 += w1[wi] * bf2f(t[pp + ch]);
      d2 += w2[wi] * bf2f(t[pp + ch + HID]);
    }
  }
  float ge = 0.5f * d1 * (1.f + erff(d1 * 0.70710678f));
  g[(long)p * HID + ch] = f2bf(ge * d2);
}

// ---------------------------------------------------------------------------
// host launcher
// ---------------------------------------------------------------------------
extern "C" void kernel_launch(void* const* d_in, const int* in_sizes, int n_in,
                              void* d_out, int out_size, void* d_ws, size_t ws_size,
                              hipStream_t stream) {
  const float* x     = (const float*)d_in[0];
  const float* mask  = (const float*)d_in[1];
  const float* edgef = (const float*)d_in[2];
  const float* ln1w  = (const float*)d_in[3];
  const float* ln1b  = (const float*)d_in[4];
  const float* Wq    = (const float*)d_in[5];
  const float* Wk    = (const float*)d_in[6];
  const float* Wv    = (const float*)d_in[7];
  const float* ln2w  = (const float*)d_in[8];
  const float* ln2b  = (const float*)d_in[9];
  const float* w_in  = (const float*)d_in[10];
  const float* w_dw  = (const float*)d_in[11];
  const float* w_out = (const float*)d_in[12];
  float* out = (float*)d_out;

  // scratch layout (t aliases the dead Aq|Akv|Q|K|V region: both 63,438,848 B)
  const size_t TC   = (size_t)NTOK * C_;
  char* ws = (char*)d_ws;
  __bf16* Aq   = (__bf16*)(ws);
  __bf16* Akv  = (__bf16*)(ws + TC * 2);
  float*  Qf   = (float*)(ws + TC * 4);
  float*  Kf   = (float*)(ws + TC * 4 + TC * 4);
  float*  Vf   = (float*)(ws + TC * 4 + TC * 8);
  __bf16* Tbuf = (__bf16*)(ws);                       // alias, 63,438,848 B
  size_t off   = TC * 4 + TC * 12;                    // 63,438,848
  float*  xres = (float*)(ws + off);                  off += TC * 4;
  __bf16* xn2  = (__bf16*)(ws + off);                 off += TC * 2;
  __bf16* gbuf = (__bf16*)(ws + off);                 off += (size_t)NTOK * HID * 2;
  __bf16* WqB  = (__bf16*)(ws + off);                 off += (size_t)C_ * C_ * 2;
  __bf16* WkB  = (__bf16*)(ws + off);                 off += (size_t)C_ * C_ * 2;
  __bf16* WvB  = (__bf16*)(ws + off);                 off += (size_t)C_ * C_ * 2;
  __bf16* WinB = (__bf16*)(ws + off);                 off += (size_t)HID2 * C_ * 2;
  __bf16* WoutB= (__bf16*)(ws + off);                 off += (size_t)C_ * HID * 2;

  // 1) weights -> bf16
  k_w2bf<<<64, 256, 0, stream>>>(Wq, WqB, C_ * C_);
  k_w2bf<<<64, 256, 0, stream>>>(Wk, WkB, C_ * C_);
  k_w2bf<<<64, 256, 0, stream>>>(Wv, WvB, C_ * C_);
  k_w2bf<<<512, 256, 0, stream>>>(w_in, WinB, HID2 * C_);
  k_w2bf<<<256, 256, 0, stream>>>(w_out, WoutB, C_ * HID);

  // 2) LN1 + bilinear + tokenize
  k_pre<<<(NPIX + 255) / 256, 256, 0, stream>>>(x, mask, edgef, ln1w, ln1b, Aq, Akv);

  // 3) Q/K/V projections (bf16 WMMA GEMM, M=30976 N=128 K=128)
  dim3 g1(NTOK / 128, 1);
  k_gemm<<<g1, 256, 0, stream>>>(Aq,  WqB, C_, C_, 0, Qf, nullptr, nullptr);
  k_gemm<<<g1, 256, 0, stream>>>(Akv, WkB, C_, C_, 0, Kf, nullptr, nullptr);
  k_gemm<<<g1, 256, 0, stream>>>(Akv, WvB, C_, C_, 0, Vf, nullptr, nullptr);

  // 4) attention core + scrambled residual
  k_attn<<<NTOK, 64, 0, stream>>>(Qf, Kf, Vf, x, xres);

  // 5) LN2 -> pixel-major bf16
  k_ln2<<<(NPIX + 255) / 256, 256, 0, stream>>>(xres, ln2w, ln2b, xn2);

  // 6) 1x1 conv 128 -> 1024 (M=30976 N=1024 K=128), bf16 out into aliased Tbuf
  dim3 g2(NTOK / 128, HID2 / 128);
  k_gemm<<<g2, 256, 0, stream>>>(xn2, WinB, C_, HID2, 1, nullptr, Tbuf, nullptr);

  // 7) depthwise 3x3 + GELU gate
  long ng = (long)NPIX * HID;
  k_dwgate<<<(unsigned)((ng + 255) / 256), 256, 0, stream>>>(Tbuf, w_dw, gbuf);

  // 8) 1x1 conv 512 -> 128 (M=30976 N=128 K=512) + transpose + residual -> out
  dim3 g3(NTOK / 128, 1);
  k_gemm<<<g3, 256, 0, stream>>>(gbuf, WoutB, HID, C_, 2, out, nullptr, xres);

  (void)in_sizes; (void)n_in; (void)out_size; (void)ws_size;
}
